// GNNEncoder_15496242004460
// MI455X (gfx1250) — matrix operand order repaired and checked by reading, not compiled, not run on hardware
//
#include <hip/hip_runtime.h>
#include <hip/hip_bf16.h>
#include <math.h>

// ---------------------------------------------------------------------------
// GNN encoder: GATConv -> BN -> ELU -> GINConv -> BN -> ELU
// N=100000 nodes, E=1.6M edges, all feature dims 128, H=8 heads x C=16.
// GEMMs via v_wmma_f32_16x16x32_f16 (wave32 CDNA5 WMMA), edge phase via
// coalesced float4 gathers + global_atomic_add_f32 scatters with scalar
// (s_load) edge-index reads.
// ---------------------------------------------------------------------------

typedef __attribute__((ext_vector_type(16))) _Float16 v16h;
typedef __attribute__((ext_vector_type(8)))  float    v8f;

#define FEAT 128
#define HEADS 8
#define CH 16
#define NEG_SLOPE 0.2f
#define BN_EPS 1e-5f

// ---------------------------------------------------------------------------
// WMMA GEMM: Out[r, :] = (escale*A0[r, :] + A1[r, :]) @ W + bias
// A0,A1: [nrows,128] f32 (A1 optional), W: [128,128] f32 row-major (K x N),
// bias optional [128]. 256 threads = 8 waves, each wave owns a 16-row tile,
// block covers 128 rows. K=128 in 4 chunks of 32, N=128 in 8 tiles of 16.
// ---------------------------------------------------------------------------
__launch_bounds__(256)
__global__ void gemm128_wmma(const float* __restrict__ A0,
                             const float* __restrict__ A1,
                             const float* __restrict__ epsp,
                             const float* __restrict__ W,
                             const float* __restrict__ bias,
                             float* __restrict__ Out,
                             int nrows)
{
    __shared__ _Float16 sW[FEAT * FEAT];   // 32 KB: weights as f16
    __shared__ _Float16 sA[8][16 * FEAT];  // 32 KB: per-wave A tiles as f16

    const int tid  = threadIdx.x;
    const int wave = tid >> 5;
    const int lane = tid & 31;

    float escale = 1.0f;
    if (epsp) escale = 1.0f + epsp[0];

    // Cooperative load of the full weight matrix into LDS (f16).
    for (int i = tid; i < FEAT * FEAT; i += 256)
        sW[i] = (_Float16)W[i];

    const int row0 = blockIdx.x * 128 + wave * 16;

    // Stage this wave's 16x128 A tile into LDS (f16); uniform branch on A1.
    if (A1) {
        for (int i = lane; i < 16 * FEAT; i += 32) {
            const int r = i >> 7, c = i & (FEAT - 1);
            int gr = row0 + r;
            if (gr >= nrows) gr = 0;
            sA[wave][i] = (_Float16)(escale * A0[(size_t)gr * FEAT + c] +
                                     A1[(size_t)gr * FEAT + c]);
        }
    } else {
        for (int i = lane; i < 16 * FEAT; i += 32) {
            const int r = i >> 7, c = i & (FEAT - 1);
            int gr = row0 + r;
            if (gr >= nrows) gr = 0;
            sA[wave][i] = (_Float16)A0[(size_t)gr * FEAT + c];
        }
    }

    // Preload per-lane bias values (8 independent loads, latency hidden
    // behind the WMMA loop; avoids per-element branch+load in the epilogue).
    const int ncol = lane & 15;
    float bv[8];
    if (bias) {
#pragma unroll
        for (int t = 0; t < 8; ++t) bv[t] = bias[t * 16 + ncol];
    } else {
#pragma unroll
        for (int t = 0; t < 8; ++t) bv[t] = 0.0f;
    }

    __syncthreads();

    v8f acc[8] = {};

    const int mrow = lane & 15;
    const int koff = (lane >> 4) ? 8 : 0;        // A layout: hi half-wave holds K+8

    for (int kc = 0; kc < 4; ++kc) {
        // A fragment: 16-bit A-matrix 16x32 VGPR layout (ISA 7.12.2).
        v16h a;
#pragma unroll
        for (int v = 0; v < 8; ++v) {
            const int klo = (v < 4) ? (2 * v) : (16 + 2 * (v - 4));
            const int k0  = kc * 32 + klo + koff;
            a[2 * v]     = sA[wave][mrow * FEAT + k0];
            a[2 * v + 1] = sA[wave][mrow * FEAT + k0 + 1];
        }
        const int krow = kc * 32 + (lane >> 4) * 8;  // B: hi half-wave holds K+8, N = lane & 15
#pragma unroll
        for (int t = 0; t < 8; ++t) {
            v16h b;
#pragma unroll
            for (int i = 0; i < 16; ++i)
                b[i] = sW[(krow + (i < 8 ? i : i + 8)) * FEAT + t * 16 + ncol];
            acc[t] = __builtin_amdgcn_wmma_f32_16x16x32_f16(
                false, a, false, b, (short)0, acc[t], false, false);
        }
    }

    // C/D layout: VGPR r -> M = r + (lane>=16 ? 8 : 0), N = lane & 15.
    const int mo = (lane >> 4) * 8;
#pragma unroll
    for (int r = 0; r < 8; ++r) {
        const int grow = row0 + mo + r;
        if (grow < nrows) {
            float* op = Out + (size_t)grow * FEAT + ncol;
#pragma unroll
            for (int t = 0; t < 8; ++t)
                op[t * 16] = acc[t][r] + bv[t];
        }
    }
}

// ---------------------------------------------------------------------------
// Attention coefficients: al_s[n,h] = <h[n,h,:], a_src[h,:]>, same for al_d.
// ---------------------------------------------------------------------------
__global__ void attn_coeff_kernel(const float* __restrict__ h,
                                  const float* __restrict__ a_src,
                                  const float* __restrict__ a_dst,
                                  float* __restrict__ al_s,
                                  float* __restrict__ al_d,
                                  int nrows)
{
    int tid = blockIdx.x * blockDim.x + threadIdx.x;
    if (tid >= nrows * HEADS) return;
    const int n  = tid >> 3;
    const int hh = tid & 7;
    const float* hp = h + (size_t)n * FEAT + hh * CH;
    const float* as = a_src + hh * CH;
    const float* ad = a_dst + hh * CH;
    float ss = 0.f, sd = 0.f;
#pragma unroll
    for (int c = 0; c < CH; ++c) {
        const float v = hp[c];
        ss += v * as[c];
        sd += v * ad[c];
    }
    al_s[tid] = ss;
    al_d[tid] = sd;
}

__device__ __forceinline__ float atomicMaxFloat(float* addr, float value)
{
    return (value >= 0.0f)
        ? __int_as_float(atomicMax((int*)addr, __float_as_int(value)))
        : __uint_as_float(atomicMin((unsigned int*)addr, __float_as_uint(value)));
}

__global__ void fill_kernel(float* __restrict__ p, float v, int n)
{
    int tid = blockIdx.x * blockDim.x + threadIdx.x;
    if (tid < n) p[tid] = v;
}

// Edge pass 1: e = leaky_relu(al_s[src]+al_d[dst]); scatter-max into m[dst].
__global__ void edge_logit_kernel(const int* __restrict__ ei, int E,
                                  const float* __restrict__ al_s,
                                  const float* __restrict__ al_d,
                                  float* __restrict__ estore,
                                  float* __restrict__ mbuf)
{
    int tid = blockIdx.x * blockDim.x + threadIdx.x;
    if (tid >= E * HEADS) return;
    const int e_id = tid >> 3;
    const int hh   = tid & 7;
    const int d    = ei[e_id];
    const int s    = ei[E + e_id];
    float v = al_s[s * HEADS + hh] + al_d[d * HEADS + hh];
    v = (v > 0.f) ? v : NEG_SLOPE * v;
    estore[tid] = v;
    atomicMaxFloat(&mbuf[d * HEADS + hh], v);
}

// Edge pass 2: ex = exp(e - m[dst]); scatter-add denominators; keep ex.
__global__ void edge_exp_kernel(const int* __restrict__ ei, int E,
                                const float* __restrict__ mbuf,
                                float* __restrict__ estore,
                                float* __restrict__ sbuf)
{
    int tid = blockIdx.x * blockDim.x + threadIdx.x;
    if (tid >= E * HEADS) return;
    const int e_id = tid >> 3;
    const int hh   = tid & 7;
    const int d    = ei[e_id];
    float mm = mbuf[d * HEADS + hh];
    if (!isfinite(mm)) mm = 0.f;
    const float ex = expf(estore[tid] - mm);
    estore[tid] = ex;
    atomicAdd(&sbuf[d * HEADS + hh], ex);
}

// Edge pass 3: out1[dst] += alpha * h[src].  One wave == one edge: lane
// carries 4 channels; edge index is wave-uniform -> force into SGPR so the
// index reads become scalar s_load instead of 32 redundant vector loads.
__global__ void edge_message_kernel(const int* __restrict__ ei, int E,
                                    const float* __restrict__ estore,
                                    const float* __restrict__ sbuf,
                                    const float* __restrict__ h,
                                    float* __restrict__ out1)
{
    int tid = blockIdx.x * blockDim.x + threadIdx.x;
    if (tid >= E * 32) return;
    const int e_id = __builtin_amdgcn_readfirstlane(tid >> 5);  // wave-uniform
    const int part = tid & 31;
    const int cb   = part * 4;
    const int hh   = cb >> 4;
    const int d    = __builtin_amdgcn_readfirstlane(ei[e_id]);
    const int s    = __builtin_amdgcn_readfirstlane(ei[E + e_id]);
    const float ex    = estore[e_id * HEADS + hh];
    const float alpha = ex / (sbuf[d * HEADS + hh] + 1e-16f);
    const float4 hv = *(const float4*)(h + (size_t)s * FEAT + cb);
    float* op = out1 + (size_t)d * FEAT + cb;
    atomicAdd(op + 0, hv.x * alpha);
    atomicAdd(op + 1, hv.y * alpha);
    atomicAdd(op + 2, hv.z * alpha);
    atomicAdd(op + 3, hv.w * alpha);
}

// GIN neighbor sum: agg[dst] += x1[src]   (same wave-per-edge structure)
__global__ void edge_agg_kernel(const int* __restrict__ ei, int E,
                                const float* __restrict__ x1,
                                float* __restrict__ agg)
{
    int tid = blockIdx.x * blockDim.x + threadIdx.x;
    if (tid >= E * 32) return;
    const int e_id = __builtin_amdgcn_readfirstlane(tid >> 5);
    const int cb   = (tid & 31) * 4;
    const int d    = __builtin_amdgcn_readfirstlane(ei[e_id]);
    const int s    = __builtin_amdgcn_readfirstlane(ei[E + e_id]);
    const float4 v = *(const float4*)(x1 + (size_t)s * FEAT + cb);
    float* op = agg + (size_t)d * FEAT + cb;
    atomicAdd(op + 0, v.x);
    atomicAdd(op + 1, v.y);
    atomicAdd(op + 2, v.z);
    atomicAdd(op + 3, v.w);
}

// Per-column sum / sum-of-squares for BatchNorm batch statistics.
__launch_bounds__(128)
__global__ void bn_stats_kernel(const float* __restrict__ Z,
                                float* __restrict__ sum,
                                float* __restrict__ sumsq,
                                int nrows)
{
    const int c   = threadIdx.x;          // 0..127
    const int rpb = (nrows + gridDim.x - 1) / gridDim.x;
    const int r0  = blockIdx.x * rpb;
    int r1 = r0 + rpb;
    if (r1 > nrows) r1 = nrows;
    float s = 0.f, q = 0.f;
    for (int r = r0; r < r1; ++r) {
        if (r + 16 < r1)  // stream-ahead hint -> global_prefetch_b8
            __builtin_prefetch(&Z[(size_t)(r + 16) * FEAT + c], 0, 1);
        const float v = Z[(size_t)r * FEAT + c];
        s += v;
        q += v * v;
    }
    atomicAdd(&sum[c], s);
    atomicAdd(&sumsq[c], q);
}

// y = elu(g*(z-mu)*rsqrt(var+eps)+beta), batch-stat BN.
__global__ void bn_elu_kernel(const float* __restrict__ Z,
                              const float* __restrict__ sum,
                              const float* __restrict__ sumsq,
                              const float* __restrict__ g,
                              const float* __restrict__ beta,
                              float* __restrict__ out,
                              int nrows)
{
    int tid = blockIdx.x * blockDim.x + threadIdx.x;
    if (tid >= nrows * FEAT) return;
    const int c = tid & (FEAT - 1);
    const float invn = 1.0f / (float)nrows;
    const float mu   = sum[c] * invn;
    const float var  = sumsq[c] * invn - mu * mu;
    const float z    = Z[tid];
    float y = g[c] * (z - mu) * rsqrtf(var + BN_EPS) + beta[c];
    out[tid] = (y > 0.f) ? y : (expf(y) - 1.0f);
}

// ---------------------------------------------------------------------------
extern "C" void kernel_launch(void* const* d_in, const int* in_sizes, int n_in,
                              void* d_out, int out_size, void* d_ws, size_t ws_size,
                              hipStream_t stream)
{
    const float* x      = (const float*)d_in[0];
    const int*   ei     = (const int*)  d_in[1];
    const float* W_gat  = (const float*)d_in[2];
    const float* a_src  = (const float*)d_in[3];
    const float* a_dst  = (const float*)d_in[4];
    // d_in[5] = b_gat (cancels through BatchNorm; zeros in setup)
    const float* g1     = (const float*)d_in[6];
    const float* beta1  = (const float*)d_in[7];
    const float* gin_e  = (const float*)d_in[8];
    const float* W_gin  = (const float*)d_in[9];
    const float* b_gin  = (const float*)d_in[10];
    const float* g2     = (const float*)d_in[11];
    const float* beta2  = (const float*)d_in[12];

    const int N = in_sizes[0] / FEAT;   // 100000
    const int E = in_sizes[1] / 2;      // 1600000

    // ---- workspace layout (f32), with aliasing to bound footprint ----
    float* ws     = (float*)d_ws;
    float* h      = ws;                      // N*128   (reused for h2 later)
    float* al_s   = h      + (size_t)N * FEAT;
    float* al_d   = al_s   + (size_t)N * HEADS;
    float* mbuf   = al_d   + (size_t)N * HEADS;
    float* sbuf   = mbuf   + (size_t)N * HEADS;
    float* estore = sbuf   + (size_t)N * HEADS;   // E*8  (reused for agg)
    float* out1   = estore + (size_t)E * HEADS;   // N*128
    float* x1     = out1   + (size_t)N * FEAT;    // N*128
    float* stats  = x1     + (size_t)N * FEAT;    // 4*128
    float* agg    = estore;                       // alias: estore dead after pass 3
    float* h2     = h;                            // alias: h dead after pass 3
    float* sum1 = stats, *sq1 = stats + FEAT, *sum2 = stats + 2 * FEAT, *sq2 = stats + 3 * FEAT;

    const int TPB = 256;
    const int gemm_blocks = (N + 127) / 128;

    // ---- init ----
    hipMemsetAsync(sbuf, 0, (size_t)N * HEADS * sizeof(float), stream);
    hipMemsetAsync(out1, 0, (size_t)N * FEAT * sizeof(float), stream);
    hipMemsetAsync(stats, 0, 4 * FEAT * sizeof(float), stream);
    fill_kernel<<<(N * HEADS + TPB - 1) / TPB, TPB, 0, stream>>>(mbuf, -INFINITY, N * HEADS);

    // ---- GAT: h = x @ W_gat (WMMA) ----
    gemm128_wmma<<<gemm_blocks, TPB, 0, stream>>>(x, nullptr, nullptr, W_gat, nullptr, h, N);

    // ---- attention coefficients ----
    attn_coeff_kernel<<<(N * HEADS + TPB - 1) / TPB, TPB, 0, stream>>>(h, a_src, a_dst, al_s, al_d, N);

    // ---- scatter softmax + message aggregation ----
    const int eh_blocks = (E * HEADS + TPB - 1) / TPB;
    edge_logit_kernel<<<eh_blocks, TPB, 0, stream>>>(ei, E, al_s, al_d, estore, mbuf);
    edge_exp_kernel<<<eh_blocks, TPB, 0, stream>>>(ei, E, mbuf, estore, sbuf);
    const int e32_blocks = (E * 32 + TPB - 1) / TPB;
    edge_message_kernel<<<e32_blocks, TPB, 0, stream>>>(ei, E, estore, sbuf, h, out1);

    // ---- BN1 + ELU -> x1 ----
    bn_stats_kernel<<<256, 128, 0, stream>>>(out1, sum1, sq1, N);
    bn_elu_kernel<<<(N * FEAT + TPB - 1) / TPB, TPB, 0, stream>>>(out1, sum1, sq1, g1, beta1, x1, N);

    // ---- GIN: agg = sum_neighbors x1 ; h2 = ((1+eps)*x1+agg) @ W_gin + b_gin ----
    hipMemsetAsync(agg, 0, (size_t)N * FEAT * sizeof(float), stream);  // after pass 3: alias safe
    edge_agg_kernel<<<e32_blocks, TPB, 0, stream>>>(ei, E, x1, agg);
    gemm128_wmma<<<gemm_blocks, TPB, 0, stream>>>(x1, agg, gin_e, W_gin, b_gin, h2, N);

    // ---- BN2 + ELU -> output ----
    bn_stats_kernel<<<256, 128, 0, stream>>>(h2, sum2, sq2, N);
    bn_elu_kernel<<<(N * FEAT + TPB - 1) / TPB, TPB, 0, stream>>>(h2, sum2, sq2, g2, beta2, (float*)d_out, N);

    (void)n_in; (void)out_size; (void)ws_size;
}
